// GATLayer_90280212562336
// MI455X (gfx1250) — matrix operand order repaired and checked by reading, not compiled, not run on hardware
//
#include <hip/hip_runtime.h>
#include <hip/hip_bf16.h>

#define NN 4096
#define IN_DIM 128
#define OD 128      // HEADS*OUT_DIM
#define HEADS 4
#define OUT_DIM 32

typedef __attribute__((ext_vector_type(16))) __bf16 v16bf;
typedef __attribute__((ext_vector_type(8)))  float  v8f;

union AB16 { v16bf v; unsigned short s[16]; unsigned d[8]; uint4 q[2]; };

__device__ __forceinline__ unsigned short f32_to_bf16_rne(float f) {
    unsigned u = __float_as_uint(f);
    u += 0x7FFFu + ((u >> 16) & 1u);
    return (unsigned short)(u >> 16);
}

// Hardware packed f32->bf16 RNE convert: D = {bf16(hi), bf16(lo)}, S0 -> D[15:0].
__device__ __forceinline__ unsigned cvt_pk_bf16(float lo, float hi) {
    unsigned r;
    asm("v_cvt_pk_bf16_f32 %0, %1, %2" : "=v"(r) : "v"(lo), "v"(hi));
    return r;
}

// ---------------------------------------------------------------------------
// Kernel 1: H = X @ W^T (f32), e_src/e_dst = <H, a_src/a_dst> per head,
// plus bf16 transposed copy Ht[c][m] for WMMA B-operand loads.
// ---------------------------------------------------------------------------
__global__ __launch_bounds__(128) void gat_prep(
    const float* __restrict__ X, const float* __restrict__ W,
    const float* __restrict__ a_src, const float* __restrict__ a_dst,
    unsigned short* __restrict__ Htb,   // [OD][NN] bf16 bits
    float* __restrict__ es,             // [HEADS][NN]
    float* __restrict__ ed)             // [HEADS][NN]
{
    __shared__ float Xs[8][IN_DIM];
    __shared__ float Hs[8][OD];
    const int t  = threadIdx.x;          // 0..127 == output column o
    const int n0 = blockIdx.x * 8;

    #pragma unroll
    for (int r = 0; r < 8; ++r) Xs[r][t] = X[(size_t)(n0 + r) * IN_DIM + t];
    __syncthreads();

    const float* Wrow = W + (size_t)t * IN_DIM;
    float acc[8];
    #pragma unroll
    for (int r = 0; r < 8; ++r) acc[r] = 0.f;
    for (int i = 0; i < IN_DIM; ++i) {
        float w = Wrow[i];
        #pragma unroll
        for (int r = 0; r < 8; ++r) acc[r] += Xs[r][i] * w;
    }
    #pragma unroll
    for (int r = 0; r < 8; ++r) {
        Hs[r][t] = acc[r];
        Htb[(size_t)t * NN + n0 + r] = f32_to_bf16_rne(acc[r]);
    }
    __syncthreads();

    if (t < 64) {
        int r = t >> 3, idx = t & 7, h = idx >> 1, isDst = idx & 1;
        const float* av = (isDst ? a_dst : a_src) + h * OUT_DIM;
        const float* hv = &Hs[r][h * OUT_DIM];
        float s = 0.f;
        #pragma unroll
        for (int d = 0; d < OUT_DIM; ++d) s += hv[d] * av[d];
        (isDst ? ed : es)[h * NN + n0 + r] = s;
    }
}

// ---------------------------------------------------------------------------
// Fused attention chunk body (32 columns). DIAG=true only for the single
// chunk that can contain the self-loop diagonal. Branchless; denominator is
// produced by a third WMMA against an all-ones B matrix (accd); bf16 weight
// packing via hardware v_cvt_pk_bf16_f32 (1 op per pair instead of ~6).
// ---------------------------------------------------------------------------
template <bool DIAG>
__device__ __forceinline__ void gat_chunk(
    int m0, int kb, int mb, int rowg, float esr,
    const float* __restrict__ Arow, const float* __restrict__ edh,
    const unsigned short* __restrict__ Hb0, const unsigned short* __restrict__ Hb1,
    v16bf ones, v8f& acc0, v8f& acc1, v8f& accd)
{
    // A (adjacency) and e_dst in exactly the A-operand K pattern
    float4 a0  = *(const float4*)(Arow + m0 + kb);
    float4 a0b = *(const float4*)(Arow + m0 + kb + 4);
    float4 a1  = *(const float4*)(Arow + m0 + kb + 16);
    float4 a1b = *(const float4*)(Arow + m0 + kb + 20);
    float4 e0  = *(const float4*)(edh + m0 + kb);
    float4 e0b = *(const float4*)(edh + m0 + kb + 4);
    float4 e1  = *(const float4*)(edh + m0 + kb + 16);
    float4 e1b = *(const float4*)(edh + m0 + kb + 20);

    AB16 b0, b1;  // H tiles, two 16B loads per lane (at +mb and +mb+16), same K pattern as A
    b0.q[0] = *(const uint4*)(Hb0 + m0 + mb);
    b0.q[1] = *(const uint4*)(Hb0 + m0 + mb + 16);
    b1.q[0] = *(const uint4*)(Hb1 + m0 + mb);
    b1.q[1] = *(const uint4*)(Hb1 + m0 + mb + 16);

    float av[16] = {a0.x,a0.y,a0.z,a0.w, a0b.x,a0b.y,a0b.z,a0b.w,
                    a1.x,a1.y,a1.z,a1.w, a1b.x,a1b.y,a1b.z,a1b.w};
    float ev[16] = {e0.x,e0.y,e0.z,e0.w, e0b.x,e0b.y,e0b.z,e0b.w,
                    e1.x,e1.y,e1.z,e1.w, e1b.x,e1b.y,e1b.z,e1b.w};

    float w[16];
    #pragma unroll
    for (int i = 0; i < 16; ++i) {
        float aval = av[i];
        if (DIAG) {
            int col = m0 + kb + ((i & 8) << 1) + (i & 7);  // global column index
            aval += (col == rowg) ? 1.0f : 0.0f;           // + identity (self loop)
        }
        float f = esr + ev[i];
        f = fmaxf(f, 0.2f * f);                            // leaky relu (==ternary)
        // w = max(A',1e-8) * exp(leaky).  The A'>0 mask is omitted: an exactly
        // zero A' (prob 2^-24/element) would contribute ~4e-10 relative, far
        // below the bf16 quantization already applied to the weights.
        w[i] = fmaxf(aval, 1e-8f) * __expf(f);
    }

    AB16 wa;
    #pragma unroll
    for (int j = 0; j < 8; ++j) wa.d[j] = cvt_pk_bf16(w[2*j], w[2*j + 1]);

    acc0 = __builtin_amdgcn_wmma_f32_16x16x32_bf16(
               false, wa.v, false, b0.v, (short)0, acc0, false, false);
    acc1 = __builtin_amdgcn_wmma_f32_16x16x32_bf16(
               false, wa.v, false, b1.v, (short)0, acc1, false, false);
    accd = __builtin_amdgcn_wmma_f32_16x16x32_bf16(
               false, wa.v, false, ones, (short)0, accd, false, false);
}

// ---------------------------------------------------------------------------
// Kernel 2: fused attention. Block = 128 threads = 4 waves; wave w = head w,
// block handles 16 rows. alpha = A'*exp(leaky(e_src+e_dst)) / rowsum, fused
// into three v_wmma_f32_16x16x32_bf16 per 32-column chunk (2 numerator tiles
// + 1 ones-matrix row-sum). Loop unrolled 2x onto independent accumulator
// sets to break WMMA C-chains and deepen the load pipeline.
// ---------------------------------------------------------------------------
__global__ __launch_bounds__(128) void gat_attn(
    const float* __restrict__ A,
    const unsigned short* __restrict__ Htb,
    const float* __restrict__ es, const float* __restrict__ ed,
    float* __restrict__ out)
{
    const int lane = threadIdx.x & 31;
    const int h    = threadIdx.x >> 5;      // wave id = head
    const int n0   = blockIdx.x * 16;
    const int row  = lane & 15;             // A-operand matrix row (M)
    const int half = lane >> 4;             // 0/1
    const int rowg = n0 + row;

    const float esr   = es[h * NN + rowg];
    const float* Arow = A + (size_t)rowg * NN;
    const float* edh  = ed + h * NN;
    const unsigned short* Hb0 = Htb + (size_t)(h * OUT_DIM + row) * NN; // col d=row
    const unsigned short* Hb1 = Hb0 + (size_t)16 * NN;                  // col d=row+16

    const int kb = half * 8;     // K base for A-operand elements 0..7
    const int mb = half * 8;     // K base for B-operand elements 0..7 (8..15 at +16)
    const int diagChunk = n0 & ~31;

    AB16 onesu;
    #pragma unroll
    for (int i = 0; i < 16; ++i) onesu.s[i] = 0x3F80;  // bf16 1.0
    const v16bf ones = onesu.v;

    v8f n0A = {}, n1A = {}, dA = {};
    v8f n0B = {}, n1B = {}, dB = {};

    int m0 = 0;
    for (; m0 + 64 <= diagChunk; m0 += 64) {
        gat_chunk<false>(m0,      kb, mb, rowg, esr, Arow, edh, Hb0, Hb1, ones, n0A, n1A, dA);
        gat_chunk<false>(m0 + 32, kb, mb, rowg, esr, Arow, edh, Hb0, Hb1, ones, n0B, n1B, dB);
    }
    if (m0 < diagChunk) {
        gat_chunk<false>(m0, kb, mb, rowg, esr, Arow, edh, Hb0, Hb1, ones, n0A, n1A, dA);
        m0 += 32;
    }
    gat_chunk<true>(m0, kb, mb, rowg, esr, Arow, edh, Hb0, Hb1, ones, n0B, n1B, dB);
    m0 += 32;
    for (; m0 + 64 <= NN; m0 += 64) {
        gat_chunk<false>(m0,      kb, mb, rowg, esr, Arow, edh, Hb0, Hb1, ones, n0A, n1A, dA);
        gat_chunk<false>(m0 + 32, kb, mb, rowg, esr, Arow, edh, Hb0, Hb1, ones, n0B, n1B, dB);
    }
    if (m0 < NN)
        gat_chunk<false>(m0, kb, mb, rowg, esr, Arow, edh, Hb0, Hb1, ones, n0A, n1A, dA);

    // Row-sum accumulator: every column of D holds the row sum, and row
    // M = r + 8*half lives in VGPR r of this very lane -> no shuffles.
    const int dcol = lane & 15;
    #pragma unroll
    for (int r = 0; r < 8; ++r) {
        int M = r + 8 * half;                 // D-matrix row for this VGPR
        float inv = 1.0f / (dA[r] + dB[r]);
        float* orow = out + (size_t)(n0 + M) * OD + h * OUT_DIM;
        orow[dcol]      = (n0A[r] + n0B[r]) * inv;
        orow[dcol + 16] = (n1A[r] + n1B[r]) * inv;
    }
}

extern "C" void kernel_launch(void* const* d_in, const int* in_sizes, int n_in,
                              void* d_out, int out_size, void* d_ws, size_t ws_size,
                              hipStream_t stream) {
    const float* X     = (const float*)d_in[0];
    const float* A     = (const float*)d_in[1];
    const float* W     = (const float*)d_in[2];
    const float* a_src = (const float*)d_in[3];
    const float* a_dst = (const float*)d_in[4];
    float* out = (float*)d_out;

    unsigned short* Htb = (unsigned short*)d_ws;                 // 128*4096*2 = 1 MB
    float* es = (float*)((char*)d_ws + (size_t)OD * NN * 2);     // 4*4096 f32
    float* ed = es + HEADS * NN;                                 // 4*4096 f32

    gat_prep<<<NN / 8, 128, 0, stream>>>(X, W, a_src, a_dst, Htb, es, ed);
    gat_attn<<<NN / 16, 128, 0, stream>>>(A, Htb, es, ed, out);
}
